// PointEncoder_180388626800
// MI455X (gfx1250) — matrix operand order repaired and checked by reading, not compiled, not run on hardware
//
#include <hip/hip_runtime.h>

// ---------------------------------------------------------------------------
// Types for CDNA5 WMMA (wave32, 16x16x32 bf16 -> f32)
// ---------------------------------------------------------------------------
typedef __attribute__((ext_vector_type(16))) __bf16 bf16x16;
typedef __attribute__((ext_vector_type(8)))  __bf16 bf16x8;
typedef __attribute__((ext_vector_type(8)))  float  f32x8;

__device__ __forceinline__ bf16x16 cat16(bf16x8 lo, bf16x8 hi) {
    return __builtin_shufflevector(lo, hi, 0,1,2,3,4,5,6,7,8,9,10,11,12,13,14,15);
}

__device__ __forceinline__ f32x8 wmma_bf16(bf16x16 a, bf16x16 b, f32x8 c) {
    // D = A(16x32 bf16) * B(32x16 bf16) + C(16x16 f32)
    return __builtin_amdgcn_wmma_f32_16x16x32_bf16(
        /*neg_a=*/false, a, /*neg_b=*/false, b,
        /*c_mod=*/(short)0, c, /*reuse_a=*/false, /*reuse_b=*/false);
}

// ---------------------------------------------------------------------------
// Weight convert + transpose: Wt[n*K + k] = (bf16) W[k*N + n]   (W is [K,N])
// ---------------------------------------------------------------------------
__global__ __launch_bounds__(256)
void convT_kernel(const float* __restrict__ W, __bf16* __restrict__ Wt,
                  int K, int N) {
    long t = (long)blockIdx.x * 256 + threadIdx.x;
    if (t >= (long)K * N) return;
    int n = (int)(t / K);
    int k = (int)(t % K);
    Wt[t] = (__bf16)W[(long)k * N + n];
}

// ---------------------------------------------------------------------------
// bf16 transpose: Vt[c*M + j] = V[j*C + c]    (V is [M,C])
// ---------------------------------------------------------------------------
__global__ __launch_bounds__(256)
void transpose_bf_kernel(const __bf16* __restrict__ V, __bf16* __restrict__ Vt,
                         int M, int C) {
    long t = (long)blockIdx.x * 256 + threadIdx.x;
    if (t >= (long)M * C) return;
    Vt[t] = V[(long)(t % M) * C + (int)(t / M)];
}

// ---------------------------------------------------------------------------
// Block reduction helper (256 threads): sum + sumsq
// ---------------------------------------------------------------------------
__device__ __forceinline__ void block_reduce2(float& s, float& q) {
    __shared__ float bs[256], bq[256];
    int tid = threadIdx.x;
    bs[tid] = s; bq[tid] = q;
    __syncthreads();
    for (int off = 128; off > 0; off >>= 1) {
        if (tid < off) { bs[tid] += bs[tid + off]; bq[tid] += bq[tid + off]; }
        __syncthreads();
    }
    s = bs[0]; q = bq[0];
    __syncthreads();
}

// ---------------------------------------------------------------------------
// FF1: h = relu(LN(x @ W1 + b1)); x[N,3], W1[3,512] fp32, h bf16 [N,512]
// one block (256 thr) per row
// ---------------------------------------------------------------------------
__global__ __launch_bounds__(256)
void ff1_kernel(const float* __restrict__ x, const float* __restrict__ W1,
                const float* __restrict__ b1, const float* __restrict__ g1,
                const float* __restrict__ be1, __bf16* __restrict__ h) {
    const int row = blockIdx.x;
    const float x0 = x[row * 3 + 0], x1 = x[row * 3 + 1], x2 = x[row * 3 + 2];
    float y[2], s = 0.f, q = 0.f;
    for (int t = 0; t < 2; ++t) {
        int c = threadIdx.x + t * 256;
        float v = fmaf(x0, W1[c], fmaf(x1, W1[512 + c], fmaf(x2, W1[1024 + c], b1[c])));
        y[t] = v; s += v; q += v * v;
    }
    block_reduce2(s, q);
    float mean = s * (1.0f / 512.0f);
    float var  = q * (1.0f / 512.0f) - mean * mean;
    float inv  = rsqrtf(var + 1e-5f);
    for (int t = 0; t < 2; ++t) {
        int c = threadIdx.x + t * 256;
        float v = (y[t] - mean) * inv * g1[c] + be1[c];
        h[(long)row * 512 + c] = (__bf16)fmaxf(v, 0.0f);
    }
}

// ---------------------------------------------------------------------------
// LayerNorm + ReLU over rows of fp32 pre-activations; writes bf16 and/or fp32
// Ncols in {512, 1024}; one block (256 thr) per row
// ---------------------------------------------------------------------------
__global__ __launch_bounds__(256)
void ln_relu_kernel(const float* __restrict__ Y, int Ncols,
                    const float* __restrict__ g, const float* __restrict__ be,
                    __bf16* __restrict__ outb, float* __restrict__ outf) {
    const int row = blockIdx.x;
    const float* y = Y + (long)row * Ncols;
    const int cnt = Ncols >> 8;   // 2 or 4
    float v[4], s = 0.f, q = 0.f;
    for (int t = 0; t < cnt; ++t) {
        int c = threadIdx.x + t * 256;
        v[t] = y[c]; s += v[t]; q += v[t] * v[t];
    }
    block_reduce2(s, q);
    float rn   = 1.0f / (float)Ncols;
    float mean = s * rn;
    float var  = q * rn - mean * mean;
    float inv  = rsqrtf(var + 1e-5f);
    for (int t = 0; t < cnt; ++t) {
        int c = threadIdx.x + t * 256;
        float o = fmaxf((v[t] - mean) * inv * g[c] + be[c], 0.0f);
        long idx = (long)row * Ncols + c;
        if (outb) outb[idx] = (__bf16)o;
        if (outf) outf[idx] = o;
    }
}

// ---------------------------------------------------------------------------
// WMMA GEMM: C[M,N] = A[M,K](bf16, lda) @ Bt[N,K](bf16, ldb)^T (+bias)
// block = 256 threads = 8 waves in a 2(M) x 4(N) grid; wave tile = 32x32
// grid = (N/128, M/64). Output fp32 (Cf) or bf16 (Cb).
// ---------------------------------------------------------------------------
__global__ __launch_bounds__(256)
void gemm_bf16_kernel(const __bf16* __restrict__ A, int lda,
                      const __bf16* __restrict__ Bt, int ldb,
                      const float* __restrict__ bias,
                      float* __restrict__ Cf, __bf16* __restrict__ Cb, int ldc,
                      int K) {
    const int lane = threadIdx.x & 31;
    const int w    = threadIdx.x >> 5;
    const int wm = w & 1, wn = w >> 1;
    const int m0 = blockIdx.y * 64  + wm * 32;
    const int n0 = blockIdx.x * 128 + wn * 32;
    const int r = lane & 15, h = lane >> 4;

    const __bf16* a0 = A  + (size_t)(m0 + r)      * lda + 16 * h;
    const __bf16* a1 = A  + (size_t)(m0 + 16 + r) * lda + 16 * h;
    const __bf16* b0 = Bt + (size_t)(n0 + r)      * ldb + 16 * h;
    const __bf16* b1 = Bt + (size_t)(n0 + 16 + r) * ldb + 16 * h;

    f32x8 acc00 = {}, acc01 = {}, acc10 = {}, acc11 = {};

    for (int k = 0; k < K; k += 32) {
        bf16x16 af0 = cat16(*(const bf16x8*)(a0 + k), *(const bf16x8*)(a0 + k + 8));
        bf16x16 af1 = cat16(*(const bf16x8*)(a1 + k), *(const bf16x8*)(a1 + k + 8));
        bf16x16 bf0 = *(const bf16x16*)(b0 + k);
        bf16x16 bf1 = *(const bf16x16*)(b1 + k);
        __builtin_prefetch(a0 + k + 256, 0, 1);
        __builtin_prefetch(b0 + k + 256, 0, 1);
        acc00 = wmma_bf16(af0, bf0, acc00);
        acc01 = wmma_bf16(af0, bf1, acc01);
        acc10 = wmma_bf16(af1, bf0, acc10);
        acc11 = wmma_bf16(af1, bf1, acc11);
    }

    f32x8 accs[4] = {acc00, acc01, acc10, acc11};
    for (int t = 0; t < 4; ++t) {
        int i = t >> 1, j = t & 1;
        int cn = n0 + j * 16 + r;
        float bv = bias ? bias[cn] : 0.0f;
        for (int v = 0; v < 8; ++v) {
            int row = m0 + i * 16 + v + 8 * h;  // C-layout: row = vgpr + 8*half
            float val = accs[t][v] + bv;
            size_t idx = (size_t)row * ldc + cn;
            if (Cf) Cf[idx] = val;
            if (Cb) Cb[idx] = (__bf16)val;
        }
    }
}

// ---------------------------------------------------------------------------
// Flash attention: O = softmax(Q K^T / sqrt(128)) @ V
// Q[M,128] bf16, Kmat[M,128] bf16, Vt[512,M] bf16 (V transposed), O[M,512] bf16
// Each wave: 16 query rows x 128 output cols; streams keys in steps of 32.
// grid = (M/16 * 4) / 8 blocks of 256 threads.
// ---------------------------------------------------------------------------
__global__ __launch_bounds__(256)
void flash_attn_kernel(const __bf16* __restrict__ Q,
                       const __bf16* __restrict__ Kmat,
                       const __bf16* __restrict__ Vt,
                       __bf16* __restrict__ O, int Mtot) {
    __shared__ __bf16 pbuf[8][16 * 32];   // per-wave P tile (16 rows x 32 keys)

    const int lane = threadIdx.x & 31;
    const int w    = threadIdx.x >> 5;
    const int task = blockIdx.x * 8 + w;
    const int m0 = (task >> 2) * 16;      // query row tile
    const int c0 = (task & 3) * 128;      // output column chunk
    const int r = lane & 15, h = lane >> 4;
    const float scale = 0.08838834764831845f;  // 1/sqrt(128)

    // Resident Q fragments (k = 0..127)
    bf16x16 qf[4];
    {
        const __bf16* qp = Q + (size_t)(m0 + r) * 128 + 16 * h;
        for (int kk = 0; kk < 4; ++kk)
            qf[kk] = cat16(*(const bf16x8*)(qp + kk * 32),
                           *(const bf16x8*)(qp + kk * 32 + 8));
    }

    f32x8 o[8] = {};                       // 8 x (16x16) output col tiles
    float mrow[8], lrow[8];
    for (int i = 0; i < 8; ++i) { mrow[i] = -3.0e38f; lrow[i] = 0.0f; }

    const __bf16* k0base = Kmat + (size_t)r * 128 + 16 * h;
    const __bf16* k1base = Kmat + (size_t)(16 + r) * 128 + 16 * h;
    const __bf16* vbase  = Vt + (size_t)(c0 + r) * Mtot + 16 * h;

    for (int j0 = 0; j0 < Mtot; j0 += 32) {
        // ---- S = Q @ K_tile^T  (two 16x16 tiles covering 32 keys) ----
        f32x8 s0 = {}, s1 = {};
        for (int kk = 0; kk < 4; ++kk) {
            bf16x16 bk0 = *(const bf16x16*)(k0base + (size_t)j0 * 128 + kk * 32);
            bf16x16 bk1 = *(const bf16x16*)(k1base + (size_t)j0 * 128 + kk * 32);
            s0 = wmma_bf16(qf[kk], bk0, s0);
            s1 = wmma_bf16(qf[kk], bk1, s1);
        }

        // ---- online softmax update (rows live across 16-lane halves) ----
        float p0[8], p1[8], corr[8];
        for (int i = 0; i < 8; ++i) {
            float t0 = s0[i] * scale, t1 = s1[i] * scale;
            float t = fmaxf(t0, t1);
            t = fmaxf(t, __shfl_xor(t, 1, 32));
            t = fmaxf(t, __shfl_xor(t, 2, 32));
            t = fmaxf(t, __shfl_xor(t, 4, 32));
            t = fmaxf(t, __shfl_xor(t, 8, 32));
            float nm = fmaxf(mrow[i], t);
            float c  = __expf(mrow[i] - nm);
            p0[i] = __expf(t0 - nm);
            p1[i] = __expf(t1 - nm);
            float rs = p0[i] + p1[i];
            rs += __shfl_xor(rs, 1, 32);
            rs += __shfl_xor(rs, 2, 32);
            rs += __shfl_xor(rs, 4, 32);
            rs += __shfl_xor(rs, 8, 32);
            lrow[i] = lrow[i] * c + rs;
            mrow[i] = nm;
            corr[i] = c;
        }
        for (int ct = 0; ct < 8; ++ct)
            for (int i = 0; i < 8; ++i)
                o[ct][i] *= corr[i];

        // ---- C-layout -> A-layout via per-wave LDS: P is 16x32 bf16 ----
        for (int i = 0; i < 8; ++i) {
            int row = i + 8 * h;
            pbuf[w][row * 32 + r]      = (__bf16)p0[i];
            pbuf[w][row * 32 + 16 + r] = (__bf16)p1[i];
        }
        bf16x16 pf;
        {
            const __bf16* pp = &pbuf[w][r * 32 + 16 * h];
            pf = cat16(*(const bf16x8*)pp, *(const bf16x8*)(pp + 8));
        }

        // ---- O += P @ V_tile  (8 column tiles of 16) ----
        for (int ct = 0; ct < 8; ++ct) {
            bf16x16 vf = *(const bf16x16*)(vbase + (size_t)ct * 16 * Mtot + j0);
            o[ct] = wmma_bf16(pf, vf, o[ct]);
        }
    }

    // ---- finalize: O /= l, store bf16 ----
    float inv[8];
    for (int i = 0; i < 8; ++i) inv[i] = 1.0f / lrow[i];
    for (int ct = 0; ct < 8; ++ct) {
        for (int i = 0; i < 8; ++i) {
            int row = m0 + i + 8 * h;
            int col = c0 + ct * 16 + r;
            O[(size_t)row * 512 + col] = (__bf16)(o[ct][i] * inv[i]);
        }
    }
}

// ---------------------------------------------------------------------------
// Host orchestration
// ---------------------------------------------------------------------------
extern "C" void kernel_launch(void* const* d_in, const int* in_sizes, int n_in,
                              void* d_out, int out_size, void* d_ws, size_t ws_size,
                              hipStream_t stream) {
    (void)in_sizes; (void)n_in; (void)out_size; (void)ws_size;

    const float* x   = (const float*)d_in[0];
    const float* W1  = (const float*)d_in[1];
    const float* b1  = (const float*)d_in[2];
    const float* g1  = (const float*)d_in[3];
    const float* be1 = (const float*)d_in[4];
    const float* W2  = (const float*)d_in[5];
    const float* b2  = (const float*)d_in[6];
    const float* g2  = (const float*)d_in[7];
    const float* be2 = (const float*)d_in[8];
    const float* WQ  = (const float*)d_in[9];
    const float* WK  = (const float*)d_in[10];
    const float* WV  = (const float*)d_in[11];
    const float* WO  = (const float*)d_in[12];
    const float* Wf  = (const float*)d_in[13];
    const float* bff = (const float*)d_in[14];
    const float* gf  = (const float*)d_in[15];
    const float* bef = (const float*)d_in[16];
    float* out = (float*)d_out;

    const int N = 8192, D = 512, MD = 128, OUTF = 1024, CATD = 2048;

    char* wp = (char*)d_ws;
    auto alloc = [&](size_t bytes) -> char* {
        char* q = wp;
        wp += (bytes + 255) & ~(size_t)255;
        return q;
    };

    __bf16* W2t = (__bf16*)alloc((size_t)D * D * 2);        // [512][512]
    __bf16* WQt = (__bf16*)alloc((size_t)MD * D * 2);       // [128][512]
    __bf16* WKt = (__bf16*)alloc((size_t)MD * D * 2);
    __bf16* WVt = (__bf16*)alloc((size_t)D * D * 2);
    __bf16* WOt = (__bf16*)alloc((size_t)D * D * 2);
    __bf16* Wft = (__bf16*)alloc((size_t)OUTF * CATD * 2);  // [1024][2048]
    __bf16* h    = (__bf16*)alloc((size_t)N * D * 2);
    float*  tmpf = (float*)alloc((size_t)N * OUTF * 4);
    __bf16* Qb   = (__bf16*)alloc((size_t)N * MD * 2);
    __bf16* Kb   = (__bf16*)alloc((size_t)N * MD * 2);
    __bf16* Vb   = (__bf16*)alloc((size_t)N * D * 2);
    __bf16* Vtb  = (__bf16*)alloc((size_t)N * D * 2);
    __bf16* AVb  = (__bf16*)alloc((size_t)N * D * 2);
    __bf16* cat  = (__bf16*)alloc((size_t)N * CATD * 2);

    auto cblocks = [](long n) { return (unsigned)((n + 255) / 256); };

    // Weight conversion (fp32 [K,N] -> bf16 [N,K])
    convT_kernel<<<cblocks((long)D * D), 256, 0, stream>>>(W2, W2t, D, D);
    convT_kernel<<<cblocks((long)D * MD), 256, 0, stream>>>(WQ, WQt, D, MD);
    convT_kernel<<<cblocks((long)D * MD), 256, 0, stream>>>(WK, WKt, D, MD);
    convT_kernel<<<cblocks((long)D * D), 256, 0, stream>>>(WV, WVt, D, D);
    convT_kernel<<<cblocks((long)D * D), 256, 0, stream>>>(WO, WOt, D, D);
    convT_kernel<<<cblocks((long)CATD * OUTF), 256, 0, stream>>>(Wf, Wft, CATD, OUTF);

    // FF1: x[N,3] -> h[N,512] bf16
    ff1_kernel<<<N, 256, 0, stream>>>(x, W1, b1, g1, be1, h);

    // FF2: h @ W2 + b2 -> LN/ReLU -> h
    gemm_bf16_kernel<<<dim3(D / 128, N / 64), 256, 0, stream>>>(
        h, D, W2t, D, b2, tmpf, nullptr, D, D);
    ln_relu_kernel<<<N, 256, 0, stream>>>(tmpf, D, g2, be2, h, nullptr);

    // 4 attention rounds
    for (int rnd = 0; rnd < 4; ++rnd) {
        const __bf16* in = (rnd == 0) ? h : (cat + (size_t)(rnd - 1) * D);
        const int lda    = (rnd == 0) ? D : CATD;

        gemm_bf16_kernel<<<dim3(MD / 128, N / 64), 256, 0, stream>>>(
            in, lda, WQt, D, nullptr, nullptr, Qb, MD, D);
        gemm_bf16_kernel<<<dim3(MD / 128, N / 64), 256, 0, stream>>>(
            in, lda, WKt, D, nullptr, nullptr, Kb, MD, D);
        gemm_bf16_kernel<<<dim3(D / 128, N / 64), 256, 0, stream>>>(
            in, lda, WVt, D, nullptr, nullptr, Vb, D, D);

        transpose_bf_kernel<<<cblocks((long)N * D), 256, 0, stream>>>(Vb, Vtb, N, D);

        flash_attn_kernel<<<(N / 16) * 4 / 8, 256, 0, stream>>>(Qb, Kb, Vtb, AVb, N);

        // (A@V) @ WO -> cat slice (also next round's input)
        gemm_bf16_kernel<<<dim3(D / 128, N / 64), 256, 0, stream>>>(
            AVb, D, WOt, D, nullptr, nullptr, cat + (size_t)rnd * D, CATD, D);
    }

    // Final FF: cat[N,2048] @ Wf + bf -> LN/ReLU -> out fp32
    gemm_bf16_kernel<<<dim3(OUTF / 128, N / 64), 256, 0, stream>>>(
        cat, CATD, Wft, CATD, bff, tmpf, nullptr, OUTF, CATD);
    ln_relu_kernel<<<N, 256, 0, stream>>>(tmpf, OUTF, gf, bef, nullptr, out);
}
